// FFTAttention_2817498546864
// MI455X (gfx1250) — matrix operand order repaired and checked by reading, not compiled, not run on hardware
//
#include <hip/hip_runtime.h>
#include <hip/hip_bf16.h>
#include <math.h>

#define D_MODEL 1024
#define N_HEADS 16
#define HEAD_DIM 64
#define SEQ 2048
#define BATCH 4
#define LOG2S 11
#define PI_F 3.14159265358979323846f

typedef __attribute__((ext_vector_type(16))) __bf16 v16bf;
typedef __attribute__((ext_vector_type(8)))  __bf16 v8bf;
typedef __attribute__((ext_vector_type(8)))  float  v8f;

// Load a 16-element bf16 fragment as two b128 chunks `gap` elements apart.
static __device__ __forceinline__ v16bf load_frag(const __bf16* __restrict__ p,
                                                  int gap)
{
    const v8bf lo = *(const v8bf*)p;
    const v8bf hi = *(const v8bf*)(p + gap);
    return __builtin_shufflevector(lo, hi,
        0, 1, 2, 3, 4, 5, 6, 7, 8, 9, 10, 11, 12, 13, 14, 15);
}

// ---------------------------------------------------------------------------
// f32 -> bf16, contiguous. Each thread converts 8 elements (2x b128 load,
// 1x b128 store).
// ---------------------------------------------------------------------------
__global__ __launch_bounds__(256) void cvt_bf16(
    const float* __restrict__ in, __bf16* __restrict__ out, int n)
{
    const int idx = (blockIdx.x * 256 + threadIdx.x) * 8;
    if (idx >= n) return;
    const float4 lo = *(const float4*)(in + idx);
    const float4 hi = *(const float4*)(in + idx + 4);
    v8bf o;
    o[0] = (__bf16)lo.x; o[1] = (__bf16)lo.y; o[2] = (__bf16)lo.z; o[3] = (__bf16)lo.w;
    o[4] = (__bf16)hi.x; o[5] = (__bf16)hi.y; o[6] = (__bf16)hi.z; o[7] = (__bf16)hi.w;
    *(v8bf*)(out + idx) = o;
}

// ---------------------------------------------------------------------------
// Transpose + convert: outT[n][k] = bf16(in[k][n]).  LDS-tiled 32x32 so both
// the global read and the global write are coalesced.  Grid: (N/32, K/32).
// ---------------------------------------------------------------------------
__global__ __launch_bounds__(256) void transpose_cvt(
    const float* __restrict__ in, __bf16* __restrict__ outT, int K, int N)
{
    __shared__ float tile[32][33];
    const int n0 = blockIdx.x * 32;
    const int k0 = blockIdx.y * 32;
    const int tx = threadIdx.x;        // 0..31
    const int ty = threadIdx.y;        // 0..7
#pragma unroll
    for (int i = 0; i < 4; ++i)
        tile[ty + 8 * i][tx] = in[(size_t)(k0 + ty + 8 * i) * N + (n0 + tx)];
    __syncthreads();
#pragma unroll
    for (int i = 0; i < 4; ++i)
        outT[(size_t)(n0 + ty + 8 * i) * K + (k0 + tx)] =
            (__bf16)tile[tx][ty + 8 * i];
}

// ---------------------------------------------------------------------------
// GEMM: C[M,N] = A[M,K] (bf16) x Bt[N,K] (bf16, pre-transposed) + bias[N].
// One wave -> 16x64 strip (1 A-frag x 4 B-frags, f32 accum).  K is a
// compile-time constant: the ping-pong unroll-by-2 removes all buffer-copy
// movs, and full unrolling folds every k-offset into the b128 load's 24-bit
// immediate (max 2 KB here), so the inner loop is loads + WMMAs only.
// ---------------------------------------------------------------------------
template <int K>
__global__ __launch_bounds__(256) void gemm_bias_wmma(
    const __bf16* __restrict__ A, const __bf16* __restrict__ Bt,
    const float* __restrict__ bias, float* __restrict__ C,
    int M, int N)
{
    const int lane = threadIdx.x & 31;
    const int wave = threadIdx.x >> 5;
    const int nTiles64 = N >> 6;
    const int tile = blockIdx.x * 8 + wave;
    const int mt = tile / nTiles64;
    const int nt = tile - mt * nTiles64;
    if (mt >= (M >> 4)) return;            // wave-uniform exit (EXEC stays full)

    const int m0   = mt << 4;
    const int n0   = nt << 6;
    const int half = lane >> 4;            // which 16-lane half of the wave
    const int l15  = lane & 15;

    v8f acc[4] = {};

    // A: row M = m0+l15; K chunk [half*16, half*16+16)
    const __bf16* __restrict__ arow = A + (size_t)(m0 + l15) * K + half * 16;
    // B: row N = n0+t*16+l15; K chunk [half*16, half*16+16)
    const __bf16* __restrict__ bptr[4];
#pragma unroll
    for (int t = 0; t < 4; ++t)
        bptr[t] = Bt + (size_t)(n0 + t * 16 + l15) * K + half * 16;

    constexpr int NSTEP = K / 32;          // 32 for K=1024 (even)

    // ---- pipeline prologue: fragments for step 0 into buffer A ----
    v16bf aA = load_frag(arow, 8);
    v16bf bA[4];
#pragma unroll
    for (int t = 0; t < 4; ++t) bA[t] = load_frag(bptr[t], 8);

    // ---- ping-pong steady state: no buffer rotation copies ----
#pragma unroll
    for (int it = 0; it < NSTEP / 2 - 1; ++it) {
        const int kB = (2 * it + 1) * 32;
        const int kA = (2 * it + 2) * 32;

        const v16bf aB = load_frag(arow + kB, 8);
        v16bf bB[4];
#pragma unroll
        for (int t = 0; t < 4; ++t) bB[t] = load_frag(bptr[t] + kB, 8);
#pragma unroll
        for (int t = 0; t < 4; ++t)
            acc[t] = __builtin_amdgcn_wmma_f32_16x16x32_bf16(
                false, aA, false, bA[t], (short)0, acc[t], false, false);

        aA = load_frag(arow + kA, 8);
#pragma unroll
        for (int t = 0; t < 4; ++t) bA[t] = load_frag(bptr[t] + kA, 8);
#pragma unroll
        for (int t = 0; t < 4; ++t)
            acc[t] = __builtin_amdgcn_wmma_f32_16x16x32_bf16(
                false, aB, false, bB[t], (short)0, acc[t], false, false);
    }

    // ---- epilogue: steps NSTEP-2 (in buffer A) and NSTEP-1 ----
    {
        const int kB = (NSTEP - 1) * 32;
        const v16bf aB = load_frag(arow + kB, 8);
        v16bf bB[4];
#pragma unroll
        for (int t = 0; t < 4; ++t) bB[t] = load_frag(bptr[t] + kB, 8);
#pragma unroll
        for (int t = 0; t < 4; ++t)
            acc[t] = __builtin_amdgcn_wmma_f32_16x16x32_bf16(
                false, aA, false, bA[t], (short)0, acc[t], false, false);
#pragma unroll
        for (int t = 0; t < 4; ++t)
            acc[t] = __builtin_amdgcn_wmma_f32_16x16x32_bf16(
                false, aB, false, bB[t], (short)0, acc[t], false, false);
    }

    // C/D layout: VGPR r holds rows (half*8 + r), col = n0 + t*16 + l15
#pragma unroll
    for (int t = 0; t < 4; ++t) {
        const int n  = n0 + t * 16 + l15;
        const float bv = bias[n];
#pragma unroll
        for (int r = 0; r < 8; ++r) {
            const int m = m0 + half * 8 + r;
            C[(size_t)m * N + n] = acc[t][r] + bv;
        }
    }
}

// ---------------------------------------------------------------------------
// Fused spectral attention core, one workgroup per (b, h, d) column:
//   load q,k column -> DIF FFT (bit-reversed out) on both ->
//   pointwise q_fft * conj(k_fft) * sigmoid(filt) ->
//   DIT inverse FFT (bit-reversed in, natural out) -> real, *SCALE/SEQ ->
//   softmax over seq -> gate by v -> write bf16 (feeds GEMM2 directly).
// Everything stays in 33 KB of LDS; no complex intermediates hit HBM.
// ---------------------------------------------------------------------------
__global__ __launch_bounds__(256) void fft_attn_fused(
    const float* __restrict__ qkv,        // [B, S, 3*D] f32
    const float* __restrict__ freq_filter,// [H]
    __bf16* __restrict__ out)             // [B, S, D] bf16
{
    __shared__ float2 Qs[SEQ];
    __shared__ float2 Ks[SEQ];
    __shared__ float  red[256];

    const int col = blockIdx.x;               // 0 .. B*H*hd-1
    const int d   = col & (HEAD_DIM - 1);
    const int h   = (col >> 6) & (N_HEADS - 1);
    const int b   = col >> 10;
    const int tid = threadIdx.x;

    const size_t rs = 3 * D_MODEL;            // row stride in qkv
    const float* __restrict__ qp = qkv + (size_t)b * SEQ * rs + h * HEAD_DIM + d;
    const float* __restrict__ kp = qp + D_MODEL;
    const float* __restrict__ vp = qp + 2 * D_MODEL;

    // ---- load columns (L2-resident: whole qkv fits in 192 MB L2) ----
#pragma unroll
    for (int i = 0; i < 8; ++i) {
        const int s = tid + 256 * i;
        Qs[s] = make_float2(qp[(size_t)s * rs], 0.0f);
        Ks[s] = make_float2(kp[(size_t)s * rs], 0.0f);
    }
    __syncthreads();

    // ---- forward DIF radix-2 FFT on Q and K (output bit-reversed) ----
    for (int lg = LOG2S - 1; lg >= 0; --lg) {
        const int   m    = 1 << lg;
        const float invm = 1.0f / (float)m;
#pragma unroll
        for (int t = 0; t < 4; ++t) {
            const int bfi = tid + 256 * t;              // butterfly 0..1023
            const int j   = bfi & (m - 1);
            const int i0  = ((bfi >> lg) << (lg + 1)) | j;
            const int i1  = i0 + m;
            float sw, cw;
            __sincosf(-PI_F * (float)j * invm, &sw, &cw);

            float2 aq = Qs[i0], bq = Qs[i1];
            Qs[i0] = make_float2(aq.x + bq.x, aq.y + bq.y);
            float dx = aq.x - bq.x, dy = aq.y - bq.y;
            Qs[i1] = make_float2(dx * cw - dy * sw, dx * sw + dy * cw);

            float2 ak = Ks[i0], bk = Ks[i1];
            Ks[i0] = make_float2(ak.x + bk.x, ak.y + bk.y);
            dx = ak.x - bk.x; dy = ak.y - bk.y;
            Ks[i1] = make_float2(dx * cw - dy * sw, dx * sw + dy * cw);
        }
        __syncthreads();
    }

    // ---- frequency-domain product (elementwise: bit-reversed order OK) ----
    const float filt = 1.0f / (1.0f + __expf(-freq_filter[h]));
#pragma unroll
    for (int i = 0; i < 8; ++i) {
        const int s = tid + 256 * i;
        const float2 q = Qs[s], k = Ks[s];
        Qs[s] = make_float2((q.x * k.x + q.y * k.y) * filt,   // q * conj(k)
                            (q.y * k.x - q.x * k.y) * filt);
    }
    __syncthreads();

    // ---- inverse DIT radix-2 FFT (bit-reversed in -> natural out) ----
    for (int lg = 0; lg < LOG2S; ++lg) {
        const int   m    = 1 << lg;
        const float invm = 1.0f / (float)m;
#pragma unroll
        for (int t = 0; t < 4; ++t) {
            const int bfi = tid + 256 * t;
            const int j   = bfi & (m - 1);
            const int i0  = ((bfi >> lg) << (lg + 1)) | j;
            const int i1  = i0 + m;
            float sw, cw;
            __sincosf(PI_F * (float)j * invm, &sw, &cw);

            const float2 a  = Qs[i0];
            const float2 bb = Qs[i1];
            const float bwx = bb.x * cw - bb.y * sw;
            const float bwy = bb.x * sw + bb.y * cw;
            Qs[i0] = make_float2(a.x + bwx, a.y + bwy);
            Qs[i1] = make_float2(a.x - bwx, a.y - bwy);
        }
        __syncthreads();
    }

    // ---- softmax over seq (SCALE=1/sqrt(64), fold in 1/SEQ ifft norm) ----
    const float sc = 0.125f / (float)SEQ;
    float vals[8];
    float lmax = -INFINITY;
#pragma unroll
    for (int i = 0; i < 8; ++i) {
        const float v = Qs[tid + 256 * i].x * sc;
        vals[i] = v;
        lmax = fmaxf(lmax, v);
    }
    red[tid] = lmax;
    __syncthreads();
    for (int off = 128; off > 0; off >>= 1) {
        if (tid < off) red[tid] = fmaxf(red[tid], red[tid + off]);
        __syncthreads();
    }
    const float gmax = red[0];
    __syncthreads();

    float lsum = 0.0f;
#pragma unroll
    for (int i = 0; i < 8; ++i) {
        const float e = __expf(vals[i] - gmax);
        vals[i] = e;
        lsum += e;
    }
    red[tid] = lsum;
    __syncthreads();
    for (int off = 128; off > 0; off >>= 1) {
        if (tid < off) red[tid] += red[tid + off];
        __syncthreads();
    }
    const float inv = 1.0f / red[0];

    // ---- gate by v, write bf16 [B,S,D] ----
    __bf16* __restrict__ op = out + (size_t)b * SEQ * D_MODEL + h * HEAD_DIM + d;
#pragma unroll
    for (int i = 0; i < 8; ++i) {
        const int s = tid + 256 * i;
        op[(size_t)s * D_MODEL] = (__bf16)(vals[i] * inv * vp[(size_t)s * rs]);
    }
}

// ---------------------------------------------------------------------------
extern "C" void kernel_launch(void* const* d_in, const int* in_sizes, int n_in,
                              void* d_out, int out_size, void* d_ws, size_t ws_size,
                              hipStream_t stream)
{
    const float* x      = (const float*)d_in[0];
    const float* qkv_w  = (const float*)d_in[1];
    const float* qkv_b  = (const float*)d_in[2];
    const float* ff     = (const float*)d_in[3];
    const float* out_w  = (const float*)d_in[4];
    const float* out_b  = (const float*)d_in[5];
    float* out  = (float*)d_out;

    const int M = BATCH * SEQ;                                   // 8192
    const size_t qkvElems = (size_t)M * 3 * D_MODEL;             // 25.2 M

    // workspace carve-up (all sizes are multiples of 4 KB)
    float*  qkv     = (float*)d_ws;                              // 96 MB f32
    __bf16* xb      = (__bf16*)(qkv + qkvElems);                 // 16 MB
    __bf16* attnb   = xb + (size_t)M * D_MODEL;                  // 16 MB
    __bf16* qkv_w_t = attnb + (size_t)M * D_MODEL;               //  6 MB
    __bf16* out_w_t = qkv_w_t + (size_t)3 * D_MODEL * D_MODEL;   //  2 MB

    // 0) one-pass operand prep: bf16 activations + transposed bf16 weights
    {
        const int n = M * D_MODEL;                               // 8.4 M elems
        cvt_bf16<<<n / (256 * 8), 256, 0, stream>>>(x, xb, n);
        transpose_cvt<<<dim3((3 * D_MODEL) / 32, D_MODEL / 32), dim3(32, 8),
                        0, stream>>>(qkv_w, qkv_w_t, D_MODEL, 3 * D_MODEL);
        transpose_cvt<<<dim3(D_MODEL / 32, D_MODEL / 32), dim3(32, 8),
                        0, stream>>>(out_w, out_w_t, D_MODEL, D_MODEL);
    }

    // 1) qkv = x @ qkv_w + qkv_b          (N = 3072)
    {
        const int waves = (M / 16) * ((3 * D_MODEL) / 64);       // 24576
        gemm_bias_wmma<D_MODEL><<<waves / 8, 256, 0, stream>>>(
            xb, qkv_w_t, qkv_b, qkv, M, 3 * D_MODEL);
    }

    // 2) fused FFT -> filter -> IFFT -> softmax -> v-gate (bf16 out)
    fft_attn_fused<<<BATCH * N_HEADS * HEAD_DIM, 256, 0, stream>>>(qkv, ff, attnb);

    // 3) out = attn @ out_w + out_b       (N = 1024)
    {
        const int waves = (M / 16) * (D_MODEL / 64);             // 8192
        gemm_bias_wmma<D_MODEL><<<waves / 8, 256, 0, stream>>>(
            attnb, out_w_t, out_b, out, M, D_MODEL);
    }
}
